// SeqModelingBlock_79809082294588
// MI455X (gfx1250) — matrix operand order repaired and checked by reading, compiled, not run on hardware
//
#include <hip/hip_runtime.h>
#include <stdint.h>

typedef __attribute__((ext_vector_type(16))) __bf16 v16bf;
typedef __attribute__((ext_vector_type(8)))  float  v8f;

#define D_     1024
#define H_     16
#define HD_    64
#define TXT_   226
#define NCH_   4
#define TCH_   1250          // TXT + 1024 video tokens per chunk
#define MQ_    5000          // NCH_ * TCH_
#define STXT_  904           // 4*226
#define SVID_  3328          // 13*256
#define STOT_  4232
#define NQT_   79            // ceil(1250/16)
#define SCALE_ 0.125f        // 1/sqrt(64)

__device__ __forceinline__ unsigned short f2bf(float f) {
  union { float f; uint32_t u; } v; v.f = f;
  uint32_t x = v.u;
  x += 0x7FFFu + ((x >> 16) & 1u);          // round-to-nearest-even
  return (unsigned short)(x >> 16);
}
__device__ __forceinline__ int iminc(int a, int b) { return a < b ? a : b; }

union FragU { v16bf v; uint32_t u[8]; };

// A fragment: 16(M) x 32(K) bf16, row-major source, K contiguous.
// lanes0-15 M=0..15 K=0..15; lanes16-31 K=16..31, 2 K/VGPR.
__device__ __forceinline__ v16bf load_frag_a(const unsigned short* base, int ld,
                                             int row0, int rowmax, int k0) {
  int lane = threadIdx.x & 31;
  int m    = lane & 15;
  int kb   = (lane & 16) >> 1;               // 0 or 8
  int row  = iminc(row0 + m, rowmax);
  const uint32_t* rp = (const uint32_t*)(base + (size_t)row * ld + k0);
  FragU r;
#pragma unroll
  for (int j = 0; j < 8; ++j) {
    r.u[j] = rp[kb + j];
  }
  return r.v;
}

// B fragment 32(K) x 16(N) from "transposed" storage [N][K] (K contiguous):
// lanes0-15 K=0..15, lanes16-31 K=16..31; lane = N column.
__device__ __forceinline__ v16bf load_frag_bt(const unsigned short* base, int ld,
                                              int row0, int rowmax, int k0) {
  int lane = threadIdx.x & 31;
  int n    = lane & 15;
  int kb2  = (lane & 16) >> 1;               // 0 or 8 dwords
  int row  = iminc(row0 + n, rowmax);
  const uint32_t* rp = (const uint32_t*)(base + (size_t)row * ld + k0);
  FragU r;
#pragma unroll
  for (int j = 0; j < 8; ++j) r.u[j] = rp[kb2 + j];
  return r.v;
}

// B fragment 32(K) x 16(N) from row-major [K][ld] (contraction along rows, strided).
__device__ __forceinline__ v16bf load_frag_b_rows(const unsigned short* base, int ld,
                                                  int row0, int rowmax, int col) {
  int lane = threadIdx.x & 31;
  int n    = (lane & 15) + col;
  int kb   = lane & 16;
  FragU r;
#pragma unroll
  for (int j = 0; j < 8; ++j) {
    int ka = iminc(row0 + kb + 2 * j,     rowmax);
    int kc = iminc(row0 + kb + 2 * j + 1, rowmax);
    uint32_t lo = base[(size_t)ka * ld + n];
    uint32_t hi = base[(size_t)kc * ld + n];
    r.u[j] = lo | (hi << 16);
  }
  return r.v;
}

__device__ __forceinline__ v8f wmma_bf16(v16bf a, v16bf b, v8f c) {
  return __builtin_amdgcn_wmma_f32_16x16x32_bf16(false, a, false, b, (short)0, c,
                                                 false, false);
}

// ---------------------------------------------------------------- GEMM
// C[M,N] f32 = A[M,K](bf16) @ Bt[N,K](bf16)^T + bias.
// Wave: 32x64 tile (2x4 WMMA micro-tile -> 8 wmma per 6 fragment loads).
// Block: 8 waves as 2x4 -> 64x256 tile. N assumed multiple of 256 here (N=1024).
__global__ __launch_bounds__(256) void gemm_bf16(const unsigned short* A,
                                                 const unsigned short* Bt,
                                                 const float* bias, float* C,
                                                 int M, int N, int K) {
  int wid  = threadIdx.x >> 5;
  int row0 = blockIdx.y * 64  + (wid >> 2) * 32;
  int col0 = blockIdx.x * 256 + (wid & 3)  * 64;
  if (row0 >= M || col0 >= N) return;        // wave-uniform
  v8f z = {0.f,0.f,0.f,0.f,0.f,0.f,0.f,0.f};
  v8f acc[2][4] = {{z,z,z,z},{z,z,z,z}};
  int lane = threadIdx.x & 31;
  for (int k0 = 0; k0 < K; k0 += 32) {
    if (k0 + 32 < K) {
      __builtin_prefetch(A  + (size_t)iminc(row0 + lane, M - 1) * K + k0 + 32, 0, 1);
      __builtin_prefetch(Bt + (size_t)iminc(col0 + lane, N - 1) * K + k0 + 32, 0, 1);
      __builtin_prefetch(Bt + (size_t)iminc(col0 + 32 + lane, N - 1) * K + k0 + 32, 0, 1);
    }
    v16bf fa0 = load_frag_a (A,  K, row0,      M - 1, k0);
    v16bf fa1 = load_frag_a (A,  K, row0 + 16, M - 1, k0);
    v16bf fb0 = load_frag_bt(Bt, K, col0,      N - 1, k0);
    v16bf fb1 = load_frag_bt(Bt, K, col0 + 16, N - 1, k0);
    v16bf fb2 = load_frag_bt(Bt, K, col0 + 32, N - 1, k0);
    v16bf fb3 = load_frag_bt(Bt, K, col0 + 48, N - 1, k0);
    acc[0][0] = wmma_bf16(fa0, fb0, acc[0][0]);
    acc[0][1] = wmma_bf16(fa0, fb1, acc[0][1]);
    acc[0][2] = wmma_bf16(fa0, fb2, acc[0][2]);
    acc[0][3] = wmma_bf16(fa0, fb3, acc[0][3]);
    acc[1][0] = wmma_bf16(fa1, fb0, acc[1][0]);
    acc[1][1] = wmma_bf16(fa1, fb1, acc[1][1]);
    acc[1][2] = wmma_bf16(fa1, fb2, acc[1][2]);
    acc[1][3] = wmma_bf16(fa1, fb3, acc[1][3]);
  }
  int half = lane >> 4, nn = lane & 15;
#pragma unroll
  for (int i = 0; i < 2; ++i)
#pragma unroll
    for (int j = 0; j < 4; ++j) {
      int col = col0 + j * 16 + nn;
      if (col >= N) continue;
      float bv = bias ? bias[col] : 0.f;
#pragma unroll
      for (int r = 0; r < 8; ++r) {
        int row = row0 + i * 16 + half * 8 + r;
        if (row < M) C[(size_t)row * N + col] = acc[i][j][r] + bv;
      }
    }
}

// ------------------------------------------------------- elementwise kernels
__global__ void transpose_bf16(const float* W, unsigned short* Wt, int K, int N) {
  int i = blockIdx.x * blockDim.x + threadIdx.x;
  if (i >= K * N) return;
  int n = i % N, k = i / N;
  Wt[(size_t)n * K + k] = f2bf(W[i]);
}

__global__ void build_cur_kernel(const float* vid, const float* txt,
                                 unsigned short* curb) {
  int i = blockIdx.x * blockDim.x + threadIdx.x;
  if (i >= MQ_ * D_) return;
  int d = i & (D_ - 1);
  int row = i >> 10;
  int c = row / TCH_, t = row % TCH_;
  float v = (t < TXT_) ? txt[((size_t)(c * TXT_ + t)) * D_ + d]
                       : vid[((size_t)(c * 768 + (t - TXT_))) * D_ + d];
  curb[i] = f2bf(v);
}

// LayerNorm (per head, HD=64) + 3D RoPE on video tokens, relayout to [c][h][T][64]
__global__ void ln_rope_split(const float* X, const float* w, const float* b,
                              unsigned short* Out, int apply) {
  int gid = blockIdx.x * blockDim.x + threadIdx.x;
  if (gid >= MQ_ * H_) return;
  int h = gid & 15, row = gid >> 4;
  int c = row / TCH_, t = row % TCH_;
  const float* x = X + (size_t)row * D_ + h * HD_;
  float v[HD_];
  if (apply) {
    float mu = 0.f;
#pragma unroll 8
    for (int d = 0; d < HD_; ++d) mu += x[d];
    mu *= (1.f / HD_);
    float var = 0.f;
#pragma unroll 8
    for (int d = 0; d < HD_; ++d) { float dd = x[d] - mu; var += dd * dd; }
    var *= (1.f / HD_);
    float rs = rsqrtf(var + 1e-6f);
#pragma unroll 8
    for (int d = 0; d < HD_; ++d) v[d] = (x[d] - mu) * rs * w[d] + b[d];
    if (t >= TXT_) {
      int p = t - TXT_;
      float fp = (float)(p >> 8);
      int rem = p & 255;
      float hp = (float)(rem >> 4), wp = (float)(rem & 15);
#pragma unroll
      for (int i = 0; i < 8; ++i) {            // DT=16 section
        float ang = fp * __powf(10000.f, -(float)(2 * i) / 16.f);
        float cs = __cosf(ang), sn = __sinf(ang);
        float x1 = v[i], x2 = v[i + 8];
        v[i] = x1 * cs - x2 * sn; v[i + 8] = x1 * sn + x2 * cs;
      }
#pragma unroll
      for (int i = 0; i < 12; ++i) {           // DHE=24 section
        float ang = hp * __powf(10000.f, -(float)(2 * i) / 24.f);
        float cs = __cosf(ang), sn = __sinf(ang);
        float x1 = v[16 + i], x2 = v[28 + i];
        v[16 + i] = x1 * cs - x2 * sn; v[28 + i] = x1 * sn + x2 * cs;
      }
#pragma unroll
      for (int i = 0; i < 12; ++i) {           // DW=24 section
        float ang = wp * __powf(10000.f, -(float)(2 * i) / 24.f);
        float cs = __cosf(ang), sn = __sinf(ang);
        float x1 = v[40 + i], x2 = v[52 + i];
        v[40 + i] = x1 * cs - x2 * sn; v[52 + i] = x1 * sn + x2 * cs;
      }
    }
  } else {
#pragma unroll 8
    for (int d = 0; d < HD_; ++d) v[d] = x[d];
  }
  unsigned short* o = Out + ((size_t)(c * H_ + h) * TCH_ + t) * HD_;
#pragma unroll 8
  for (int d = 0; d < HD_; ++d) o[d] = f2bf(v[d]);
}

// --------------------------------------------------------------- attention
// One wave = one (chunk, head, 16-row q tile); flash-style online softmax.
__global__ __launch_bounds__(256) void attn_kernel(const unsigned short* Qh,
                                                   const unsigned short* Kh,
                                                   const unsigned short* Vh,
                                                   unsigned short* O) {
  __shared__ unsigned short pstage[8][16 * 32];
  int wid  = threadIdx.x >> 5;
  int lane = threadIdx.x & 31;
  int task = blockIdx.x * 8 + wid;
  if (task >= NCH_ * H_ * NQT_) return;      // wave-uniform
  int qt = task % NQT_;
  int ch = task / NQT_;
  int h = ch & 15, c = ch >> 4;
  size_t base = (size_t)(c * H_ + h) * TCH_ * HD_;
  const unsigned short* Qb = Qh + base;
  const unsigned short* Kb = Kh + base;
  const unsigned short* Vb = Vh + base;
  int q0 = qt * 16;
  v16bf qa0 = load_frag_a(Qb, HD_, q0, TCH_ - 1, 0);
  v16bf qa1 = load_frag_a(Qb, HD_, q0, TCH_ - 1, 32);
  v8f z = {0.f,0.f,0.f,0.f,0.f,0.f,0.f,0.f};
  v8f acc0 = z, acc1 = z, acc2 = z, acc3 = z;
  float m[8], l[8];
#pragma unroll
  for (int r = 0; r < 8; ++r) { m[r] = -1e30f; l[r] = 0.f; }
  int half = lane >> 4, nq = lane & 15;
  unsigned short* ps = &pstage[wid][0];

  for (int kt = 0; kt < 1280; kt += 32) {
    v8f s0 = z, s1 = z;
    v16bf kf;
    kf = load_frag_bt(Kb, HD_, kt,      TCH_ - 1, 0);  s0 = wmma_bf16(qa0, kf, s0);
    kf = load_frag_bt(Kb, HD_, kt,      TCH_ - 1, 32); s0 = wmma_bf16(qa1, kf, s0);
    kf = load_frag_bt(Kb, HD_, kt + 16, TCH_ - 1, 0);  s1 = wmma_bf16(qa0, kf, s1);
    kf = load_frag_bt(Kb, HD_, kt + 16, TCH_ - 1, 32); s1 = wmma_bf16(qa1, kf, s1);

    bool v0 = (kt + nq) < TCH_;
    bool v1 = (kt + 16 + nq) < TCH_;
    float mloc[8];
#pragma unroll
    for (int r = 0; r < 8; ++r) {
      float x0 = v0 ? s0[r] * SCALE_ : -1e30f;
      float x1 = v1 ? s1[r] * SCALE_ : -1e30f;
      s0[r] = x0; s1[r] = x1;
      mloc[r] = fmaxf(x0, x1);
    }
#pragma unroll
    for (int mk = 1; mk < 16; mk <<= 1)
#pragma unroll
      for (int r = 0; r < 8; ++r)
        mloc[r] = fmaxf(mloc[r], __shfl_xor(mloc[r], mk, 32));
    float corr[8], psum[8];
#pragma unroll
    for (int r = 0; r < 8; ++r) {
      float mn = fmaxf(m[r], mloc[r]);
      corr[r] = __expf(m[r] - mn);
      m[r] = mn;
      float p0 = __expf(s0[r] - mn), p1 = __expf(s1[r] - mn);
      s0[r] = p0; s1[r] = p1;
      psum[r] = p0 + p1;
    }
#pragma unroll
    for (int mk = 1; mk < 16; mk <<= 1)
#pragma unroll
      for (int r = 0; r < 8; ++r) psum[r] += __shfl_xor(psum[r], mk, 32);
#pragma unroll
    for (int r = 0; r < 8; ++r) {
      l[r] = l[r] * corr[r] + psum[r];
      acc0[r] *= corr[r]; acc1[r] *= corr[r];
      acc2[r] *= corr[r]; acc3[r] *= corr[r];
    }
    // stage P (16x32 bf16) in LDS so it can be re-read in A-fragment layout
#pragma unroll
    for (int r = 0; r < 8; ++r) {
      ps[(half * 8 + r) * 32 + nq]      = f2bf(s0[r]);
      ps[(half * 8 + r) * 32 + 16 + nq] = f2bf(s1[r]);
    }
    v16bf pa = load_frag_a(ps, 32, 0, 15, 0);
    v16bf vb;
    vb = load_frag_b_rows(Vb, HD_, kt, TCH_ - 1, 0);  acc0 = wmma_bf16(pa, vb, acc0);
    vb = load_frag_b_rows(Vb, HD_, kt, TCH_ - 1, 16); acc1 = wmma_bf16(pa, vb, acc1);
    vb = load_frag_b_rows(Vb, HD_, kt, TCH_ - 1, 32); acc2 = wmma_bf16(pa, vb, acc2);
    vb = load_frag_b_rows(Vb, HD_, kt, TCH_ - 1, 48); acc3 = wmma_bf16(pa, vb, acc3);
  }
  v8f accs[4] = {acc0, acc1, acc2, acc3};
#pragma unroll
  for (int f = 0; f < 4; ++f) {
    int col = h * HD_ + f * 16 + nq;
#pragma unroll
    for (int r = 0; r < 8; ++r) {
      int row = q0 + half * 8 + r;
      if (row < TCH_)
        O[((size_t)(c * TCH_ + row)) * D_ + col] = f2bf(accs[f][r] / l[r]);
    }
  }
}

// ------------------------------------------------------------------ SSM etc.
__global__ void unchunk_kernel(const float* proj, float* emb, unsigned short* embb) {
  int i = blockIdx.x * blockDim.x + threadIdx.x;
  if (i >= STOT_ * D_) return;
  int d = i & (D_ - 1), s = i >> 10;
  float val;
  if (s < STXT_) {
    int c = s / TXT_, r = s % TXT_;
    val = proj[((size_t)(c * TCH_ + r)) * D_ + d];
  } else {
    int p = s - STXT_;
    float sum = 0.f; int cnt = 0;
#pragma unroll
    for (int c = 0; c < 4; ++c) {
      int off = p - c * 768;
      if (off >= 0 && off < 1024) { sum += proj[((size_t)(c * TCH_ + TXT_ + off)) * D_ + d]; ++cnt; }
    }
    val = sum / (float)cnt;
  }
  emb[i] = val;
  embb[i] = f2bf(val);
}

__global__ void ssm_scan(const float* u, const float* gate, unsigned short* hb,
                         int T, int Dm) {
  int d = blockIdx.x * blockDim.x + threadIdx.x;
  if (d >= Dm) return;
  float a = 1.f / (1.f + __expf(-gate[d]));
  float h = 0.f;
  for (int t = 0; t < T; ++t) {
    h = a * h + u[(size_t)t * Dm + d];
    hb[(size_t)t * Dm + d] = f2bf(h);
  }
}

__device__ __forceinline__ int seqmap(int s) {
  if (s < STXT_) { int c = s / TXT_, r = s % TXT_; return (3 - c) * TXT_ + r; }
  return STXT_ + (SVID_ - 1 - (s - STXT_));
}

__global__ void combine1_kernel(const float* emb, const float* y, const float* fgt,
                                const float* fgv, float* emb2, unsigned short* revb) {
  int i = blockIdx.x * blockDim.x + threadIdx.x;
  if (i >= STOT_ * D_) return;
  int d = i & (D_ - 1), s = i >> 10;
  float g = tanhf((s < STXT_) ? fgt[d] : fgv[d]);
  float e2 = emb[i] + g * y[i];
  emb2[i] = e2;
  revb[((size_t)seqmap(s)) * D_ + d] = f2bf(e2);   // map is an involution
}

__global__ void final_kernel(const float* emb2, const float* y2r, const float* bgt,
                             const float* bgv, float* out) {
  int i = blockIdx.x * blockDim.x + threadIdx.x;
  if (i >= STOT_ * D_) return;
  int d = i & (D_ - 1), s = i >> 10;
  float y2 = y2r[((size_t)seqmap(s)) * D_ + d];
  float g = tanhf((s < STXT_) ? bgt[d] : bgv[d]);
  float o = emb2[i] + g * y2;
  size_t orow = (s < STXT_) ? (size_t)(SVID_ + s) : (size_t)(s - STXT_); // vid first
  out[orow * D_ + d] = o;
}

// ------------------------------------------------------------------- driver
extern "C" void kernel_launch(void* const* d_in, const int* in_sizes, int n_in,
                              void* d_out, int out_size, void* d_ws, size_t ws_size,
                              hipStream_t stream) {
  const float* vid  = (const float*)d_in[0];
  const float* txt  = (const float*)d_in[1];
  const float* Wq   = (const float*)d_in[2];  const float* bq = (const float*)d_in[3];
  const float* Wk   = (const float*)d_in[4];  const float* bk = (const float*)d_in[5];
  const float* Wv   = (const float*)d_in[6];  const float* bv = (const float*)d_in[7];
  const float* Wo   = (const float*)d_in[8];  const float* bo = (const float*)d_in[9];
  const float* qnw  = (const float*)d_in[10]; const float* qnb = (const float*)d_in[11];
  const float* knw  = (const float*)d_in[12]; const float* knb = (const float*)d_in[13];
  const float* Win  = (const float*)d_in[14]; const float* Wout = (const float*)d_in[15];
  const float* gate = (const float*)d_in[16];
  const float* fgt  = (const float*)d_in[17]; const float* fgv = (const float*)d_in[18];
  const float* bgt  = (const float*)d_in[19]; const float* bgv = (const float*)d_in[20];
  (void)in_sizes; (void)n_in; (void)out_size; (void)ws_size;

  uint8_t* ws = (uint8_t*)d_ws;
  size_t off = 0;
  auto alloc = [&](size_t bytes) -> void* {
    void* p = ws + off;
    off += (bytes + 255) & ~(size_t)255;
    return p;
  };
  unsigned short* WT[6];
  for (int i = 0; i < 6; ++i) WT[i] = (unsigned short*)alloc((size_t)D_ * D_ * 2);
  unsigned short* curb = (unsigned short*)alloc((size_t)MQ_ * D_ * 2);
  float* qf = (float*)alloc((size_t)MQ_ * D_ * 4);
  float* kf = (float*)alloc((size_t)MQ_ * D_ * 4);
  float* vf = (float*)alloc((size_t)MQ_ * D_ * 4);
  unsigned short* qh = (unsigned short*)alloc((size_t)MQ_ * D_ * 2);
  unsigned short* kh = (unsigned short*)alloc((size_t)MQ_ * D_ * 2);
  unsigned short* vh = (unsigned short*)alloc((size_t)MQ_ * D_ * 2);
  unsigned short* embb = (unsigned short*)alloc((size_t)STOT_ * D_ * 2);
  unsigned short* hb   = (unsigned short*)alloc((size_t)STOT_ * D_ * 2);
  unsigned short* revb = (unsigned short*)alloc((size_t)STOT_ * D_ * 2);
  float* yb   = (float*)alloc((size_t)STOT_ * D_ * 4);
  float* emb2 = (float*)alloc((size_t)STOT_ * D_ * 4);
  // aliases (lifetimes do not overlap)
  unsigned short* attnb = curb;
  float* proj = qf;
  float* emb  = kf;
  float* ub   = vf;

  const float* Wsrc[6] = {Wq, Wk, Wv, Wo, Win, Wout};
  for (int i = 0; i < 6; ++i)
    transpose_bf16<<<(D_ * D_ + 255) / 256, 256, 0, stream>>>(Wsrc[i], WT[i], D_, D_);
  build_cur_kernel<<<(MQ_ * D_ + 255) / 256, 256, 0, stream>>>(vid, txt, curb);

  dim3 gq(D_ / 256, (MQ_ + 63) / 64);
  gemm_bf16<<<gq, 256, 0, stream>>>(curb, WT[0], bq, qf, MQ_, D_, D_);
  gemm_bf16<<<gq, 256, 0, stream>>>(curb, WT[1], bk, kf, MQ_, D_, D_);
  gemm_bf16<<<gq, 256, 0, stream>>>(curb, WT[2], bv, vf, MQ_, D_, D_);

  int lrN = MQ_ * H_;
  ln_rope_split<<<(lrN + 255) / 256, 256, 0, stream>>>(qf, qnw, qnb, qh, 1);
  ln_rope_split<<<(lrN + 255) / 256, 256, 0, stream>>>(kf, knw, knb, kh, 1);
  ln_rope_split<<<(lrN + 255) / 256, 256, 0, stream>>>(vf, (const float*)nullptr,
                                                      (const float*)nullptr, vh, 0);

  int tasks = NCH_ * H_ * NQT_;
  attn_kernel<<<(tasks + 7) / 8, 256, 0, stream>>>(qh, kh, vh, attnb);

  gemm_bf16<<<gq, 256, 0, stream>>>(attnb, WT[3], bo, proj, MQ_, D_, D_);
  unchunk_kernel<<<(STOT_ * D_ + 255) / 256, 256, 0, stream>>>(proj, emb, embb);

  dim3 gs(D_ / 256, (STOT_ + 63) / 64);
  gemm_bf16<<<gs, 256, 0, stream>>>(embb, WT[4], (const float*)nullptr, ub, STOT_, D_, D_);
  ssm_scan<<<(D_ + 255) / 256, 256, 0, stream>>>(ub, gate, hb, STOT_, D_);
  gemm_bf16<<<gs, 256, 0, stream>>>(hb, WT[5], (const float*)nullptr, yb, STOT_, D_, D_);
  combine1_kernel<<<(STOT_ * D_ + 255) / 256, 256, 0, stream>>>(emb, yb, fgt, fgv,
                                                                emb2, revb);
  gemm_bf16<<<gs, 256, 0, stream>>>(revb, WT[4], (const float*)nullptr, ub, STOT_, D_, D_);
  ssm_scan<<<(D_ + 255) / 256, 256, 0, stream>>>(ub, gate, hb, STOT_, D_);
  gemm_bf16<<<gs, 256, 0, stream>>>(hb, WT[5], (const float*)nullptr, yb, STOT_, D_, D_);
  final_kernel<<<(STOT_ * D_ + 255) / 256, 256, 0, stream>>>(emb2, yb, bgt, bgv,
                                                             (float*)d_out);
}
